// HPG_12317966205116
// MI455X (gfx1250) — matrix operand order repaired and checked by reading, not compiled, not run on hardware
//
#include <hip/hip_runtime.h>
#include <hip/hip_bf16.h>

// ---------------------------------------------------------------------------
// Types for CDNA5 WMMA
// ---------------------------------------------------------------------------
typedef __attribute__((ext_vector_type(16))) __bf16          v16bf;
typedef __attribute__((ext_vector_type(16))) unsigned short  v16u;
typedef __attribute__((ext_vector_type(8)))  float           v8f;

union ABf { v16u u; v16bf b; };

static __device__ __forceinline__ unsigned short f2bf(float f) {
    unsigned u = __float_as_uint(f);
    u += 0x7FFFu + ((u >> 16) & 1u);   // round-to-nearest-even
    return (unsigned short)(u >> 16);
}

#define NCOLS 160   // 8 fused projection matrices x H(20)
#define NT    10    // N tiles of 16
#define MT    128   // rows per block (8 waves x 16)

// ---------------------------------------------------------------------------
// Weight pack: 8 (din x 20) f32 matrices -> bf16 [ktile][n][kk] (kk in 0..31),
// zero-padded in K.  Also concatenates the 8 bias vectors.
// ---------------------------------------------------------------------------
struct WB8 { const float* W[8]; const float* B[8]; };

__global__ __launch_bounds__(256)
void pack_weights(WB8 p, int din, int ktiles,
                  unsigned short* __restrict__ Wc, float* __restrict__ bc)
{
    int t = blockIdx.x * blockDim.x + threadIdx.x;
    int total = ktiles * (NCOLS * 32);
    if (t < total) {
        int kt  = t / (NCOLS * 32);
        int rem = t % (NCOLS * 32);
        int n   = rem / 32;
        int kk  = rem % 32;
        int k   = kt * 32 + kk;
        int g = n / 20, j = n % 20;
        float wv = (k < din) ? p.W[g][(size_t)k * 20 + j] : 0.0f;
        Wc[t] = f2bf(wv);
    }
    if (t < NCOLS) {
        int g = t / 20, j = t % 20;
        bc[t] = p.B[g][j];
    }
}

// ---------------------------------------------------------------------------
// WMMA GEMM:  Y[nrows x 160] (f32) = A[nrows x Kdim] (f32, cvt->bf16) @ Wc + bias
// ---------------------------------------------------------------------------
__global__ __launch_bounds__(256)
void gemm_wmma(const float* __restrict__ A, int lda, int Kdim, int ktiles,
               const unsigned short* __restrict__ Wc,
               const float* __restrict__ bias,
               float* __restrict__ Y, int nrows)
{
    __shared__ __align__(32) unsigned short Blds[NCOLS * 32];  // [n][k] layout

    int tid  = threadIdx.x;
    int wave = tid >> 5;
    int lane = tid & 31;

    int mbase = blockIdx.x * MT + wave * 16;
    int lrow  = lane & 15;
    int khalf = (lane >> 4) * 8;        // A-frag K offset per ISA layout
    int arow  = mbase + lrow;
    if (arow >= nrows) arow = nrows - 1;          // clamp, stores are guarded
    const float* __restrict__ Arow = A + (size_t)arow * lda;

    int coln = lane & 15;
    int kb16 = (lane >> 4) * 16;        // unused; B-frag K offset is khalf

    v8f acc[NT];
#pragma unroll
    for (int nt = 0; nt < NT; ++nt)
#pragma unroll
        for (int i = 0; i < 8; ++i) acc[nt][i] = 0.0f;

    for (int kt = 0; kt < ktiles; ++kt) {
        int k0 = kt * 32;

        __syncthreads();
        {
            const uint2* sp = (const uint2*)(Wc + (size_t)kt * (NCOLS * 32));
            uint2* dp = (uint2*)Blds;
#pragma unroll
            for (int i = 0; i < 5; ++i) dp[tid + 256 * i] = sp[tid + 256 * i];
        }
        __syncthreads();

        v16u au;
        if (k0 + 32 <= Kdim) {
            const float* p = Arow + k0 + khalf;
            float4 f0 = *(const float4*)(p);
            float4 f1 = *(const float4*)(p + 4);
            float4 f2 = *(const float4*)(p + 16);
            float4 f3 = *(const float4*)(p + 20);
            au[0]  = f2bf(f0.x); au[1]  = f2bf(f0.y); au[2]  = f2bf(f0.z); au[3]  = f2bf(f0.w);
            au[4]  = f2bf(f1.x); au[5]  = f2bf(f1.y); au[6]  = f2bf(f1.z); au[7]  = f2bf(f1.w);
            au[8]  = f2bf(f2.x); au[9]  = f2bf(f2.y); au[10] = f2bf(f2.z); au[11] = f2bf(f2.w);
            au[12] = f2bf(f3.x); au[13] = f2bf(f3.y); au[14] = f2bf(f3.z); au[15] = f2bf(f3.w);
        } else {
#pragma unroll
            for (int j = 0; j < 8; ++j) {
                int ka = k0 + khalf + j;
                int kb = k0 + 16 + khalf + j;
                au[j]     = (ka < Kdim) ? f2bf(Arow[ka]) : (unsigned short)0;
                au[8 + j] = (kb < Kdim) ? f2bf(Arow[kb]) : (unsigned short)0;
            }
        }
        ABf af; af.u = au;

        if (k0 + 32 < Kdim) __builtin_prefetch(Arow + k0 + 32 + khalf, 0, 1);

#pragma unroll
        for (int nt = 0; nt < NT; ++nt) {
            int col = nt * 16 + coln;
            ABf bfv;
            __builtin_memcpy(&bfv.u,
                (const unsigned short*)__builtin_assume_aligned(&Blds[col * 32 + khalf], 16),
                16);
            __builtin_memcpy((char*)&bfv.u + 16,
                (const unsigned short*)__builtin_assume_aligned(&Blds[col * 32 + 16 + khalf], 16),
                16);
            acc[nt] = __builtin_amdgcn_wmma_f32_16x16x32_bf16(
                false, af.b, false, bfv.b, (short)0, acc[nt], false, false);
        }
    }

    int rofs = (lane >> 4) * 8;
#pragma unroll
    for (int nt = 0; nt < NT; ++nt) {
        int col = nt * 16 + coln;
        float bv = bias[col];
#pragma unroll
        for (int r = 0; r < 8; ++r) {
            int row = mbase + rofs + r;
            if (row < nrows) Y[(size_t)row * NCOLS + col] = acc[nt][r] + bv;
        }
    }
}

// ---------------------------------------------------------------------------
// CSR construction (edge sets are layer-invariant: build once, reuse 4x)
// ---------------------------------------------------------------------------
__global__ __launch_bounds__(256)
void zero_int(int* __restrict__ p, int n)
{
    int t = blockIdx.x * blockDim.x + threadIdx.x;
    if (t < n) p[t] = 0;
}

__global__ __launch_bounds__(256)
void count_deg(const int* __restrict__ dst, int E, int* __restrict__ cnt)
{
    int e = blockIdx.x * blockDim.x + threadIdx.x;
    if (e < E) atomicAdd(&cnt[dst[e]], 1);
}

// per-block inclusive scan
__global__ __launch_bounds__(256)
void scan1(const int* __restrict__ cnt, int n,
           int* __restrict__ incl, int* __restrict__ bsum)
{
    __shared__ int sh[256];
    int i = blockIdx.x * 256 + threadIdx.x;
    int v = (i < n) ? cnt[i] : 0;
    sh[threadIdx.x] = v;
    __syncthreads();
#pragma unroll
    for (int d = 1; d < 256; d <<= 1) {
        int t = (threadIdx.x >= d) ? sh[threadIdx.x - d] : 0;
        __syncthreads();
        sh[threadIdx.x] += t;
        __syncthreads();
    }
    if (i < n) incl[i] = sh[threadIdx.x];
    if (threadIdx.x == 255) bsum[blockIdx.x] = sh[255];
}

// exclusive scan of block sums (nb <= 256), in place
__global__ __launch_bounds__(256)
void scan2(int* __restrict__ bsum, int nb)
{
    __shared__ int sh[256];
    int v = (threadIdx.x < nb) ? bsum[threadIdx.x] : 0;
    sh[threadIdx.x] = v;
    __syncthreads();
#pragma unroll
    for (int d = 1; d < 256; d <<= 1) {
        int t = (threadIdx.x >= d) ? sh[threadIdx.x - d] : 0;
        __syncthreads();
        sh[threadIdx.x] += t;
        __syncthreads();
    }
    if (threadIdx.x < nb) bsum[threadIdx.x] = sh[threadIdx.x] - v;
}

__global__ __launch_bounds__(256)
void scan3(const int* __restrict__ incl, const int* __restrict__ cnt,
           const int* __restrict__ bsum, int n, int* __restrict__ rs)
{
    int i = blockIdx.x * 256 + threadIdx.x;
    if (i < n) {
        int off = bsum[i >> 8];
        rs[i] = incl[i] - cnt[i] + off;
        if (i == n - 1) rs[n] = incl[i] + off;
    }
}

__global__ __launch_bounds__(256)
void copy_int(const int* __restrict__ a, int* __restrict__ b, int n)
{
    int t = blockIdx.x * blockDim.x + threadIdx.x;
    if (t < n) b[t] = a[t];
}

__global__ __launch_bounds__(256)
void csr_fill(const int* __restrict__ src, const int* __restrict__ dst, int E,
              int* __restrict__ cursor, int* __restrict__ col)
{
    int e = blockIdx.x * blockDim.x + threadIdx.x;
    if (e >= E) return;
    int p = atomicAdd(&cursor[dst[e]], 1);
    col[p] = src[e];
}

// ---------------------------------------------------------------------------
// Fused per-node attention (both edge sets) + mix + BN(eval) + leaky-relu.
// 4 lanes cooperate per node; online softmax -> zero atomics.
// ---------------------------------------------------------------------------
static __device__ __forceinline__ void attn_one(
    const float* __restrict__ Y, int off,
    const int* __restrict__ rs, const int* __restrict__ cols,
    int node, int sl, float out[5])
{
    const float* yn = Y + (size_t)node * NCOLS + off;
    float4 q[5];
#pragma unroll
    for (int i = 0; i < 5; ++i) q[i] = ((const float4*)yn)[i];

    int beg = rs[node], end = rs[node + 1];

    float m = -3.402823466e38f;   // -FLT_MAX identity (NaN-free merges)
    float den = 0.0f;
    float acc[20];
#pragma unroll
    for (int h = 0; h < 20; ++h) acc[h] = 0.0f;

    for (int j = beg + sl; j < end; j += 4) {
        int sn = cols[j];
        const float* ys = Y + (size_t)sn * NCOLS + off;
        const float4* kk = (const float4*)(ys + 20);
        float s = 0.0f;
#pragma unroll
        for (int i = 0; i < 5; ++i) {
            float4 kv = kk[i];
            s += q[i].x * kv.x + q[i].y * kv.y + q[i].z * kv.z + q[i].w * kv.w;
        }
        s *= 0.22360679774997896f;          // 1/sqrt(20)
        float Mn = fmaxf(m, s);
        float sc = __expf(m - Mn);          // first iter: exp(-huge) = 0
        float w  = __expf(s - Mn);
        den = den * sc + w;
        const float4* vv = (const float4*)(ys + 40);
#pragma unroll
        for (int i = 0; i < 5; ++i) {
            float4 v4 = vv[i];
            acc[4 * i + 0] = acc[4 * i + 0] * sc + w * v4.x;
            acc[4 * i + 1] = acc[4 * i + 1] * sc + w * v4.y;
            acc[4 * i + 2] = acc[4 * i + 2] * sc + w * v4.z;
            acc[4 * i + 3] = acc[4 * i + 3] * sc + w * v4.w;
        }
        m = Mn;
    }

    // merge (m, den, acc) across the 4 cooperating lanes with rescaling
#pragma unroll
    for (int d = 1; d < 4; d <<= 1) {
        float m2   = __shfl_xor(m, d, 32);
        float den2 = __shfl_xor(den, d, 32);
        float Mn = fmaxf(m, m2);
        float sa = __expf(m - Mn);
        float sb = __expf(m2 - Mn);
        den = den * sa + den2 * sb;
#pragma unroll
        for (int h = 0; h < 20; ++h) {
            float a2 = __shfl_xor(acc[h], d, 32);
            acc[h] = acc[h] * sa + a2 * sb;
        }
        m = Mn;
    }

    float inv = (den > 0.0f) ? 1.0f / den : 0.0f;
#pragma unroll
    for (int i = 0; i < 5; ++i) {
        int h = sl * 5 + i;
        out[i] = acc[h] * inv + yn[60 + h];   // + skip (Ws cols at off+60..79)
    }
}

__global__ __launch_bounds__(256)
void node_attn(const float* __restrict__ Y,
               const int* __restrict__ rsS, const int* __restrict__ colS,
               const int* __restrict__ rsD, const int* __restrict__ colD,
               const float* __restrict__ w1, const float* __restrict__ w2,
               const float* __restrict__ bg, const float* __restrict__ bb,
               const float* __restrict__ bm, const float* __restrict__ bv,
               int l, int n, float* __restrict__ x, float* __restrict__ fc)
{
    int t = blockIdx.x * blockDim.x + threadIdx.x;
    int node = t >> 2;
    int sl = t & 3;
    if (node >= n) return;

    float r1[5], r2[5];
    attn_one(Y, 0,  rsS, colS, node, sl, r1);   // same edges (cols 0..79)
    attn_one(Y, 80, rsD, colD, node, sl, r2);   // diff edges (cols 80..159)

    float a = w1[l], b = w2[l], sw = a + b;
#pragma unroll
    for (int i = 0; i < 5; ++i) {
        int h = sl * 5 + i;
        float xv = (a / sw) * r1[i] + (b / sw) * r2[i];
        int c = l * 20 + h;
        xv = (xv - bm[c]) * rsqrtf(bv[c] + 1e-5f);
        xv = xv * bg[c] + bb[c];
        xv = (xv >= 0.0f) ? xv : 0.01f * xv;
        x[(size_t)node * 20 + h]  = xv;
        fc[(size_t)node * 80 + c] = xv;
    }
}

// ---------------------------------------------------------------------------
// Output head: out = fc[n,80] @ Wout[80,2] + bout
// ---------------------------------------------------------------------------
__global__ __launch_bounds__(256)
void final_out(const float* __restrict__ fc, const float* __restrict__ Wout,
               const float* __restrict__ bout, float* __restrict__ out, int n)
{
    int node = blockIdx.x * blockDim.x + threadIdx.x;
    if (node >= n) return;
    const float* f = fc + (size_t)node * 80;
    float o0 = bout[0], o1 = bout[1];
#pragma unroll
    for (int i = 0; i < 80; ++i) {
        float v = f[i];
        o0 += v * Wout[i * 2 + 0];
        o1 += v * Wout[i * 2 + 1];
    }
    out[node * 2 + 0] = o0;
    out[node * 2 + 1] = o1;
}

// ---------------------------------------------------------------------------
// Host-side orchestration
// ---------------------------------------------------------------------------
extern "C" void kernel_launch(void* const* d_in, const int* in_sizes, int n_in,
                              void* d_out, int out_size, void* d_ws, size_t ws_size,
                              hipStream_t stream)
{
    (void)n_in; (void)out_size; (void)ws_size;

    const float* features = (const float*)d_in[0];
    const int*   same_idx = (const int*)d_in[1];
    const int*   diff_idx = (const int*)d_in[2];
    const int N = in_sizes[0] / 1800;
    const int E = in_sizes[1] / 2;

    const float* bn_g = (const float*)d_in[67];
    const float* bn_b = (const float*)d_in[68];
    const float* bn_m = (const float*)d_in[69];
    const float* bn_v = (const float*)d_in[70];
    const float* w1   = (const float*)d_in[71];
    const float* w2   = (const float*)d_in[72];
    const float* Wout = (const float*)d_in[73];
    const float* bout = (const float*)d_in[74];

    // Carve workspace
    char* base = (char*)d_ws;
    size_t off = 0;
    auto carve = [&](size_t bytes) -> void* {
        void* p = base + off;
        off += (bytes + 255) & ~(size_t)255;
        return p;
    };
    float*          Ybuf = (float*)carve((size_t)N * NCOLS * 4);
    unsigned short* Wc0  = (unsigned short*)carve((size_t)57 * NCOLS * 32 * 2);
    unsigned short* WcL  = (unsigned short*)carve((size_t)3 * NCOLS * 32 * 2);
    float*          bc   = (float*)carve((size_t)4 * NCOLS * 4);
    float*          xbuf = (float*)carve((size_t)N * 20 * 4);
    float*          fc   = (float*)carve((size_t)N * 80 * 4);
    int*            rsS  = (int*)carve((size_t)(N + 1) * 4);
    int*            colS = (int*)carve((size_t)E * 4);
    int*            rsD  = (int*)carve((size_t)(N + 1) * 4);
    int*            colD = (int*)carve((size_t)E * 4);
    int*            cnt  = (int*)carve((size_t)N * 4);
    int*            incl = (int*)carve((size_t)N * 4);
    int*            curs = (int*)carve((size_t)N * 4);
    int*            bsum = (int*)carve(256 * 4);

    const int nblk = (N + 255) / 256;          // 196 blocks (<= 256)
    const int eblk = (E + 255) / 256;

    // --- Build CSR for both edge sets (reused by all 4 layers) ---
    for (int setI = 0; setI < 2; ++setI) {
        const int* idx = setI == 0 ? same_idx : diff_idx;
        int* rs  = setI == 0 ? rsS : rsD;
        int* col = setI == 0 ? colS : colD;
        const int* src = idx;
        const int* dst = idx + E;

        zero_int <<<nblk, 256, 0, stream>>>(cnt, N);
        count_deg<<<eblk, 256, 0, stream>>>(dst, E, cnt);
        scan1    <<<nblk, 256, 0, stream>>>(cnt, N, incl, bsum);
        scan2    <<<1,    256, 0, stream>>>(bsum, nblk);
        scan3    <<<nblk, 256, 0, stream>>>(incl, cnt, bsum, N, rs);
        copy_int <<<nblk, 256, 0, stream>>>(rs, curs, N);
        csr_fill <<<eblk, 256, 0, stream>>>(src, dst, E, curs, col);
    }

    // --- Pack all layer weights into fused bf16 [K x 160] matrices ---
    for (int l = 0; l < 4; ++l) {
        WB8 p;
        int pbase = 3 + l * 16;
        for (int g = 0; g < 4; ++g) {
            p.W[g]     = (const float*)d_in[pbase + 2 * g];
            p.B[g]     = (const float*)d_in[pbase + 2 * g + 1];
            p.W[4 + g] = (const float*)d_in[pbase + 8 + 2 * g];
            p.B[4 + g] = (const float*)d_in[pbase + 8 + 2 * g + 1];
        }
        int din = (l == 0) ? 1800 : 20;
        int kt  = (l == 0) ? 57 : 1;
        unsigned short* wc = (l == 0) ? Wc0 : (WcL + (size_t)(l - 1) * NCOLS * 32);
        int tot = kt * NCOLS * 32;
        pack_weights<<<(tot + 255) / 256, 256, 0, stream>>>(p, din, kt, wc, bc + l * NCOLS);
    }

    const int mtiles = (N + MT - 1) / MT;

    for (int l = 0; l < 4; ++l) {
        const float* Ain = (l == 0) ? features : xbuf;
        int Kdim = (l == 0) ? 1800 : 20;
        int kt   = (l == 0) ? 57 : 1;
        const unsigned short* wc = (l == 0) ? Wc0 : (WcL + (size_t)(l - 1) * NCOLS * 32);

        gemm_wmma<<<mtiles, 256, 0, stream>>>(Ain, Kdim, Kdim, kt, wc,
                                              bc + l * NCOLS, Ybuf, N);

        node_attn<<<(N * 4 + 255) / 256, 256, 0, stream>>>(
            Ybuf, rsS, colS, rsD, colD,
            w1, w2, bn_g, bn_b, bn_m, bn_v, l, N, xbuf, fc);
    }

    final_out<<<(N + 255) / 256, 256, 0, stream>>>(fc, Wout, bout, (float*)d_out, N);
}
